// FullCA_910533067502
// MI455X (gfx1250) — matrix operand order repaired and checked by reading, not compiled, not run on hardware
//
#include <hip/hip_runtime.h>
#include <hip/hip_bf16.h>
#include <cstdint>

// ---------------------------------------------------------------------------
// FullCA on MI455X (gfx1250, wave32, WMMA).
// Pipeline:
//   1) k_downsample  (s=1..5): z_s = bilinear /2^s of x (== 2x2 avg at the
//      align_corners=False sample points:  i0 = 2^s*i + 2^(s-1) - 1).
//   2) k_perceive    (s=1..5): p_s = [z, sobelx, sobely, laplacian] (48ch),
//      replicate-padded cross-correlation, at scale resolution.
//   3) k_accum_y: y = (perceive(x) + sum_s upsample(p_s)) / 6.  The whole
//      p pyramid (67 MB) fits in the 192 MB L2, so the 20 bilinear taps per
//      channel hit L2; HBM traffic is ~write-y + read-x + read-pyramid-once.
//   4) k_mlp: fused 50->96(relu)->12 per-pixel MLP via v_wmma_f32_16x16x32_f16.
//      A-operands are staged in LDS pre-swizzled into contiguous-k fragment order
//      so each wave loads fragments with ds_load_b128 instead of u16 gathers.
// ---------------------------------------------------------------------------

#define BN 4
#define CC 12
#define HH 512
#define WW 512
#define HWP (HH * WW)

typedef __attribute__((ext_vector_type(16))) _Float16 v16h;
typedef __attribute__((ext_vector_type(8)))  float    v8f;

// 16-bit A 16x32 layout (contiguous, matches B): lane = ((k>>4)&1)<<4 | m,
// element e = k & 15   (k = K index within chunk)
__device__ __forceinline__ int afrag_elem(int kl) {
  return kl & 15;
}
__device__ __forceinline__ int afrag_lane(int kl, int m) {
  return (((kl >> 4) & 1) << 4) | m;
}

// ---------------------------------------------------------------- kernel 1
__global__ void k_downsample(const float* __restrict__ x, float* __restrict__ z,
                             int Hs, int Ws, int s) {
  int idx = blockIdx.x * blockDim.x + threadIdx.x;
  int total = BN * CC * Hs * Ws;
  if (idx >= total) return;
  int j  = idx % Ws;
  int t  = idx / Ws;
  int i  = t % Hs;
  int bc = t / Hs;
  // src coord = (i+0.5)*2^s - 0.5 -> floor = 2^s*i + 2^(s-1) - 1, frac = 0.5
  int i0 = (i << s) + (1 << (s - 1)) - 1;
  int j0 = (j << s) + (1 << (s - 1)) - 1;
  const float* xp = x + (size_t)bc * HWP;
  float v = 0.25f * (xp[(size_t)i0 * WW + j0]     + xp[(size_t)i0 * WW + j0 + 1] +
                     xp[(size_t)(i0+1) * WW + j0] + xp[(size_t)(i0+1) * WW + j0 + 1]);
  z[idx] = v;
}

// 3x3 cross-correlation taps with replicate padding (shared by k2/k3)
__device__ __forceinline__ void stencil3x3(const float* __restrict__ plane,
                                           int Hs, int Ws, int i, int j,
                                           float& c0, float& sx, float& sy, float& lap) {
  int im = i > 0 ? i - 1 : 0;
  int ip = i < Hs - 1 ? i + 1 : Hs - 1;
  int jm = j > 0 ? j - 1 : 0;
  int jp = j < Ws - 1 ? j + 1 : Ws - 1;
  float a00 = plane[(size_t)im * Ws + jm], a01 = plane[(size_t)im * Ws + j], a02 = plane[(size_t)im * Ws + jp];
  float a10 = plane[(size_t)i  * Ws + jm], a11 = plane[(size_t)i  * Ws + j], a12 = plane[(size_t)i  * Ws + jp];
  float a20 = plane[(size_t)ip * Ws + jm], a21 = plane[(size_t)ip * Ws + j], a22 = plane[(size_t)ip * Ws + jp];
  c0  = a11;
  sx  = (a02 - a00) + 2.f * (a12 - a10) + (a22 - a20);
  sy  = (a20 - a00) + 2.f * (a21 - a01) + (a22 - a02);
  lap = a00 + 2.f*a01 + a02 + 2.f*a10 - 12.f*a11 + 2.f*a12 + a20 + 2.f*a21 + a22;
}

// ---------------------------------------------------------------- kernel 2
__global__ void k_perceive(const float* __restrict__ z, float* __restrict__ p,
                           int Hs, int Ws) {
  int idx = blockIdx.x * blockDim.x + threadIdx.x;
  int total = BN * CC * Hs * Ws;
  if (idx >= total) return;
  int j = idx % Ws;
  int t = idx / Ws;
  int i = t % Hs;
  int c = (t / Hs) % CC;
  int b = t / (Hs * CC);
  size_t plane = (size_t)Hs * Ws;
  const float* zp = z + ((size_t)b * CC + c) * plane;
  float c0, sx, sy, lap;
  stencil3x3(zp, Hs, Ws, i, j, c0, sx, sy, lap);
  size_t pix = (size_t)i * Ws + j;
  float* pb = p + ((size_t)b * 48) * plane + pix;
  pb[(size_t)(c)      * plane] = c0;
  pb[(size_t)(12 + c) * plane] = sx;
  pb[(size_t)(24 + c) * plane] = sy;
  pb[(size_t)(36 + c) * plane] = lap;
}

// ---------------------------------------------------------------- kernel 3
__global__ void k_accum_y(const float* __restrict__ x,
                          const float* __restrict__ p1, const float* __restrict__ p2,
                          const float* __restrict__ p3, const float* __restrict__ p4,
                          const float* __restrict__ p5,
                          float* __restrict__ y) {
  int idx = blockIdx.x * blockDim.x + threadIdx.x;
  int total = BN * CC * HH * WW;
  if (idx >= total) return;
  int j = idx % WW;
  int t = idx / WW;
  int i = t % HH;
  int c = (t / HH) % CC;
  int b = t / (HH * CC);

  // scale-0 perception computed in-register from x
  float acc[4];
  {
    const float* xp = x + ((size_t)b * CC + c) * HWP;
    stencil3x3(xp, HH, WW, i, j, acc[0], acc[1], acc[2], acc[3]);
  }

  const float* ps[5] = {p1, p2, p3, p4, p5};
#pragma unroll
  for (int s = 1; s <= 5; ++s) {
    const int Hs = HH >> s, Ws = WW >> s;
    const float inv = 1.f / (float)(1 << s);
    float fi = (i + 0.5f) * inv - 0.5f;
    float fj = (j + 0.5f) * inv - 0.5f;
    float i0f = floorf(fi), j0f = floorf(fj);
    float wi = fi - i0f, wj = fj - j0f;
    int i0 = (int)i0f, j0 = (int)j0f;
    int i0c = min(max(i0, 0), Hs - 1), i1c = min(max(i0 + 1, 0), Hs - 1);
    int j0c = min(max(j0, 0), Ws - 1), j1c = min(max(j0 + 1, 0), Ws - 1);
    size_t plane = (size_t)Hs * Ws;
    const float* pbase = ps[s - 1] + ((size_t)b * 48 + c) * plane;
#pragma unroll
    for (int f = 0; f < 4; ++f) {
      const float* pp = pbase + (size_t)f * 12 * plane;  // channel k = f*12 + c
      float v00 = pp[(size_t)i0c * Ws + j0c], v01 = pp[(size_t)i0c * Ws + j1c];
      float v10 = pp[(size_t)i1c * Ws + j0c], v11 = pp[(size_t)i1c * Ws + j1c];
      acc[f] += (1.f - wi) * ((1.f - wj) * v00 + wj * v01) +
                wi        * ((1.f - wj) * v10 + wj * v11);
    }
  }
  size_t pix = (size_t)i * WW + j;
  float* yb = y + ((size_t)b * 48) * HWP + pix;
  const float inv6 = 1.f / 6.f;
#pragma unroll
  for (int f = 0; f < 4; ++f)
    yb[(size_t)(f * 12 + c) * HWP] = acc[f] * inv6;
}

// ---------------------------------------------------------------- kernel 4
// Per block: 128 consecutive pixels (one image row segment), 8 waves x 16 px.
// Layer1: A[16x64 f16] x B[64x96] via 6 N-tiles * 2 chunks of wmma_f32_16x16x32_f16.
// Layer2: A[16x96] x B[96x16] via 3 chunks. C->A transpose via per-wave LDS,
// scattered into contiguous-k A-fragment order so reads are ds_load_b128.
__launch_bounds__(256)
__global__ void k_mlp(const float* __restrict__ y,  const float* __restrict__ x,
                      const float* __restrict__ w1w, const float* __restrict__ w1b,
                      const float* __restrict__ w2w, const float* __restrict__ w2b,
                      const float* __restrict__ mask, float* __restrict__ out) {
  __shared__ v16h     aF[8][2][32];     // 16 KB : layer-1 A fragments, pre-swizzled
  __shared__ _Float16 w1h[96][64];      // 12 KB : layer1 weights [N][K] (K padded)
  __shared__ _Float16 w2h[16][96];      //  3 KB : layer2 weights [N padded 16][K]
  __shared__ v16h     hA[8][3][32];     // 24 KB : layer-2 A fragments (hidden act)
  __shared__ float    updL[8][16][12];  //  6 KB : per-wave update [pixel][ch]

  const int tid  = threadIdx.x;
  const int wave = tid >> 5;
  const int lane = tid & 31;
  const size_t pixBase  = (size_t)blockIdx.x * 128;
  const int b        = (int)(pixBase / HWP);
  const int pixInImg = (int)(pixBase % HWP);
  const int row0     = pixInImg / WW;   // constant: 128 | 512
  const int col0     = pixInImg % WW;

  // CDNA5 prefetch (global_prefetch_b8) for the residual/mask streams
  __builtin_prefetch(&x[((size_t)b * CC) * HWP + pixInImg], 0, 1);
  __builtin_prefetch(&mask[(size_t)b * HWP + pixInImg], 0, 1);

  // ---- stage weights into LDS (f32 -> f16), K zero-padded
  for (int idx = tid; idx < 96 * 64; idx += 256) {
    int o = idx >> 6, k = idx & 63;
    w1h[o][k] = (k < 50) ? (_Float16)w1w[o * 50 + k] : (_Float16)0.f;
  }
  for (int idx = tid; idx < 16 * 96; idx += 256) {
    int n = idx / 96, k = idx % 96;
    w2h[n][k] = (n < 12) ? (_Float16)w2w[n * 96 + k] : (_Float16)0.f;
  }
  // ---- stage the y tile directly into A-fragment order;
  //      coordinate embedding folded into K=48,49; K 50..63 zero-padded
  const float xs = (2.f * row0 + 1.f) / (float)HH - 1.f;
  for (int idx = tid; idx < 64 * 128; idx += 256) {
    int k = idx >> 7, p = idx & 127;
    float v;
    if (k < 48)       v = y[((size_t)b * 48 + k) * HWP + pixInImg + p];
    else if (k == 48) v = xs;
    else if (k == 49) v = (2.f * (float)(col0 + p) + 1.f) / (float)WW - 1.f;
    else              v = 0.f;
    int wv = p >> 4, m = p & 15;
    int chunk = k >> 5, kl = k & 31;
    ((_Float16*)&aF[wv][chunk][afrag_lane(kl, m)])[afrag_elem(kl)] = (_Float16)v;
  }
  __syncthreads();

  const int m     = lane & 15;   // pixel within wave group / N within tile
  const int half  = lane >> 4;
  const int pwave = wave * 16;

  // ---- layer-1 A fragments: straight vector loads (2x ds_load_b128 each)
  const v16h a0 = aF[wave][0][lane];
  const v16h a1 = aF[wave][1][lane];

  // ---- layer 1: 6 N-tiles of 16, K = 64 (2 WMMA chunks each)
#pragma unroll
  for (int t = 0; t < 6; ++t) {
    const int n = t * 16 + m;
    // B (contiguous order, matches A): lane n=l%16, k = half*16 + e per chunk -> contiguous
    const v16h b0 = *(const v16h*)&w1h[n][half << 4];
    const v16h b1 = *(const v16h*)&w1h[n][32 + (half << 4)];
    v8f c = {};
    c = __builtin_amdgcn_wmma_f32_16x16x32_f16(false, a0, false, b0, (short)0, c, false, false);
    c = __builtin_amdgcn_wmma_f32_16x16x32_f16(false, a1, false, b1, (short)0, c, false, false);
    // C layout: vgpr r -> M = r + 8*half, N = m.  bias + relu, scatter into the
    // layer-2 A-fragment slot for (k2 = n, m2 = pixel).
    const float bias  = w1b[n];
    const int   chunk = n >> 5;
    const int   kl    = n & 31;
    const int   el    = afrag_elem(kl);
    const int   lhalf = ((kl >> 4) & 1) << 4;
#pragma unroll
    for (int r = 0; r < 8; ++r) {
      int mm = r + (half << 3);
      float hv = c[r] + bias;
      ((_Float16*)&hA[wave][chunk][lhalf | mm])[el] = (_Float16)(hv > 0.f ? hv : 0.f);
    }
  }

  // ---- layer 2: K = 96 (3 chunks), N = 12 padded to 16
  v8f c2 = {};
#pragma unroll
  for (int cc = 0; cc < 3; ++cc) {
    const v16h a2 = hA[wave][cc][lane];                          // 2x ds_load_b128
    const v16h b2 = *(const v16h*)&w2h[m][cc * 32 + (half << 4)]; // contiguous
    c2 = __builtin_amdgcn_wmma_f32_16x16x32_f16(false, a2, false, b2, (short)0, c2, false, false);
  }

  // ---- stage update through LDS, then coalesced masked residual writes
  const float bias2 = (m < 12) ? w2b[m] : 0.f;
#pragma unroll
  for (int r = 0; r < 8; ++r) {
    int mm = r + (half << 3);
    if (m < 12) updL[wave][mm][m] = c2[r] + bias2;
  }
#pragma unroll
  for (int c0 = 0; c0 < 12; c0 += 2) {
    int ch = c0 + half;           // 2 channels x 16 pixels per iteration
    int pp = m;
    size_t gp = (size_t)pixInImg + pwave + pp;
    float u  = updL[wave][pp][ch];
    float mk = mask[(size_t)b * HWP + gp];
    size_t oi = ((size_t)b * CC + ch) * HWP + gp;
    out[oi] = x[oi] + u * mk;
  }
}

// ---------------------------------------------------------------- launcher
extern "C" void kernel_launch(void* const* d_in, const int* in_sizes, int n_in,
                              void* d_out, int out_size, void* d_ws, size_t ws_size,
                              hipStream_t stream) {
  (void)in_sizes; (void)n_in; (void)out_size; (void)ws_size;
  const float* x    = (const float*)d_in[0];
  const float* w1w  = (const float*)d_in[1];
  const float* w1b  = (const float*)d_in[2];
  const float* w2w  = (const float*)d_in[3];
  const float* w2b  = (const float*)d_in[4];
  const float* msk  = (const float*)d_in[5];
  float* out = (float*)d_out;
  float* ws  = (float*)d_ws;

  // workspace layout (floats)
  size_t off = 0;
  float* yv = ws + off;               off += (size_t)BN * 48 * HWP;           // 201 MB
  float* z[6];  float* pp[6];
  for (int s = 1; s <= 5; ++s) { z[s]  = ws + off; off += (size_t)BN * CC * (HWP >> (2 * s)); }
  for (int s = 1; s <= 5; ++s) { pp[s] = ws + off; off += (size_t)BN * 48 * (HWP >> (2 * s)); }

  for (int s = 1; s <= 5; ++s) {
    int Hs = HH >> s, Ws2 = WW >> s;
    int total = BN * CC * Hs * Ws2;
    k_downsample<<<(total + 255) / 256, 256, 0, stream>>>(x, z[s], Hs, Ws2, s);
  }
  for (int s = 1; s <= 5; ++s) {
    int Hs = HH >> s, Ws2 = WW >> s;
    int total = BN * CC * Hs * Ws2;
    k_perceive<<<(total + 255) / 256, 256, 0, stream>>>(z[s], pp[s], Hs, Ws2);
  }
  {
    int total = BN * CC * HH * WW;
    k_accum_y<<<(total + 255) / 256, 256, 0, stream>>>(x, pp[1], pp[2], pp[3], pp[4], pp[5], yv);
  }
  {
    int blocks = (BN * HWP) / 128;   // 8192, exact
    k_mlp<<<blocks, 256, 0, stream>>>(yv, x, w1w, w1b, w2w, w2b, msk, out);
  }
}
